// AttentionFlowLayer_72791105733336
// MI455X (gfx1250) — matrix operand order repaired and checked by reading, not compiled, not run on hardware
//
#include <hip/hip_runtime.h>

#define BB 32
#define TC 2048
#define TQ 256
#define DD 256

typedef __attribute__((ext_vector_type(16))) __bf16 bf16x16;
typedef __attribute__((ext_vector_type(8)))  __bf16 bf16x8;
typedef __attribute__((ext_vector_type(8)))  float  floatx8;

// ---------------------------------------------------------------------------
// Kernel 0: per-batch query prep: colconst[q] = q . w2 ; Qb (bf16 row-major),
// QT (bf16 transposed, [b][d][q]) for WMMA B-fragments.
// grid: B * (TQ/32) blocks, 256 threads
// ---------------------------------------------------------------------------
__global__ __launch_bounds__(256)
void prep_q_kernel(const float* __restrict__ query, const float* __restrict__ w,
                   float* __restrict__ colconst,
                   unsigned short* __restrict__ Qb_, unsigned short* __restrict__ QT_) {
    __bf16* Qb = reinterpret_cast<__bf16*>(Qb_);
    __bf16* QT = reinterpret_cast<__bf16*>(QT_);
    const int blk = blockIdx.x;
    const int b   = blk >> 3;
    const int q0  = (blk & 7) * 32;
    const int tid = threadIdx.x;

    __shared__ float w2s[DD];
    __shared__ float qs[32][DD + 1];
    __shared__ float red8[32][8];

    w2s[tid] = w[DD + tid];
    for (int r = 0; r < 32; ++r) {
        float v = query[((size_t)b * TQ + q0 + r) * DD + tid];
        qs[r][tid] = v;
        __bf16 hv = (__bf16)v;
        Qb[((size_t)b * TQ + q0 + r) * DD + tid] = hv;
        QT[((size_t)b * DD + tid) * TQ + q0 + r] = hv;
    }
    __syncthreads();
    {
        int r = tid >> 3, sg = tid & 7;
        float p = 0.f;
        #pragma unroll 8
        for (int i = 0; i < 32; ++i) p += qs[r][sg * 32 + i] * w2s[sg * 32 + i];
        red8[r][sg] = p;
    }
    __syncthreads();
    if (tid < 32) {
        float s = 0.f;
        #pragma unroll
        for (int i = 0; i < 8; ++i) s += red8[tid][i];
        colconst[b * TQ + q0 + tid] = s;
    }
}

// ---------------------------------------------------------------------------
// Kernel 1: main tile kernel. Block = (batch b, 64 context rows).
// 8 waves: wave w -> rows (w>>1)*16..+15, cols (w&1)*128..+127 (8 N-tiles).
// GEMM1: S = (C*w3) @ Q^T  (bf16 WMMA, f32 acc) + colconst
// softmax rows -> P (bf16 in LDS) ; rowmax(+rowconst) -> ws
// GEMM2: U = P @ Q ; epilogue streams G[:, :, 0:768] = [C, U, C*U]
// ---------------------------------------------------------------------------
__global__ __launch_bounds__(256)
void attn_tile_kernel(const float* __restrict__ ctx, const float* __restrict__ w,
                      const float* __restrict__ colconst,
                      const unsigned short* __restrict__ Qb_,
                      const unsigned short* __restrict__ QT_,
                      float* __restrict__ rowmax_out, float* __restrict__ G) {
    const __bf16* Qb = reinterpret_cast<const __bf16*>(Qb_);
    const __bf16* QT = reinterpret_cast<const __bf16*>(QT_);

    const int b   = blockIdx.y;
    const int t0  = blockIdx.x * 64;
    const int tid = threadIdx.x;
    const int lane = tid & 31;
    const int wv   = tid >> 5;      // 0..7
    const int wm   = wv >> 1;       // 0..3 : 16-row strip
    const int wn   = wv & 1;        // 0..1 : 128-col half
    const int m0   = wm * 16;
    const int n0   = wn * 128;
    const bool hi  = lane >= 16;
    const int l16  = lane & 15;

    __shared__ float w3s[DD];
    __shared__ float ccq[TQ];
    __shared__ float rowcs[64];
    __shared__ float rc4[64][4];
    __shared__ float rmaxW[2][64];
    __shared__ float rsumW[2][64];
    __shared__ union {
        __bf16 P[64][272];          // bf16 probs, 16B-aligned rows (544B stride)
        float  U[32][260];          // f32 U staging (half-tile at a time)
    } sb;

    // ---- prologue: w3, colconst, rowconst (c . w1) ----
    w3s[tid] = w[2 * DD + tid];
    ccq[tid] = colconst[b * TQ + tid];
    {
        int r = tid >> 2, sg = tid & 3;
        const float* crow = ctx + ((size_t)b * TC + t0 + r) * DD + sg * 64;
        float p = 0.f;
        #pragma unroll 8
        for (int i = 0; i < 64; ++i) p += crow[i] * w[sg * 64 + i];
        rc4[r][sg] = p;
    }
    __syncthreads();
    if (tid < 64) rowcs[tid] = rc4[tid][0] + rc4[tid][1] + rc4[tid][2] + rc4[tid][3];
    __syncthreads();

    // ---- GEMM1: S tiles ----
    floatx8 Sacc[8];
    #pragma unroll
    for (int j = 0; j < 8; ++j)
        #pragma unroll
        for (int i = 0; i < 8; ++i) Sacc[j][i] = 0.f;

    const float* arow = ctx + ((size_t)b * TC + t0 + m0 + l16) * DD;
    for (int k0 = 0; k0 < DD; k0 += 32) {
        // A fragment: 16 consecutive K halves kb..kb+15
        const int kb = k0 + (hi ? 16 : 0);
        bf16x16 a;
        #pragma unroll
        for (int i = 0; i < 8; ++i) {
            a[i]     = (__bf16)(arow[kb + i]      * w3s[kb + i]);
            a[i + 8] = (__bf16)(arow[kb + 8 + i] * w3s[kb + 8 + i]);
        }
        #pragma unroll
        for (int j = 0; j < 8; ++j) {
            // B fragment: lane holds column q = n0+j*16+l16, 16 consecutive K halves
            const __bf16* qrow = Qb + ((size_t)b * TQ + n0 + j * 16 + l16) * DD
                                    + k0 + (hi ? 16 : 0);
            bf16x16 bf = *(const bf16x16*)qrow;
            Sacc[j] = __builtin_amdgcn_wmma_f32_16x16x32_bf16(
                false, a, false, bf, (short)0, Sacc[j], false, false);
        }
    }

    // ---- add per-column constant (q . w2) ----
    #pragma unroll
    for (int j = 0; j < 8; ++j) {
        float cq = ccq[n0 + j * 16 + l16];
        #pragma unroll
        for (int r = 0; r < 8; ++r) Sacc[j][r] += cq;
    }

    // ---- row max over this wave's 128 cols (shfl within 16-lane halves) ----
    const int rowbase = m0 + (hi ? 8 : 0);
    float rmax[8];
    #pragma unroll
    for (int r = 0; r < 8; ++r) {
        float m = -3.0e38f;
        #pragma unroll
        for (int j = 0; j < 8; ++j) m = fmaxf(m, Sacc[j][r]);
        #pragma unroll
        for (int off = 1; off < 16; off <<= 1) m = fmaxf(m, __shfl_xor(m, off, 32));
        rmax[r] = m;
    }
    if (l16 == 0) {
        #pragma unroll
        for (int r = 0; r < 8; ++r) rmaxW[wn][rowbase + r] = rmax[r];
    }
    __syncthreads();
    float rmaxf[8];
    #pragma unroll
    for (int r = 0; r < 8; ++r)
        rmaxf[r] = fmaxf(rmaxW[0][rowbase + r], rmaxW[1][rowbase + r]);
    if (wn == 0 && l16 == 0) {  // row max of full S (incl. row const) for b_t path
        #pragma unroll
        for (int r = 0; r < 8; ++r)
            rowmax_out[(size_t)b * TC + t0 + rowbase + r] = rmaxf[r] + rowcs[rowbase + r];
    }

    // ---- exp + row sum ----
    float rsum[8];
    #pragma unroll
    for (int r = 0; r < 8; ++r) {
        float s = 0.f;
        #pragma unroll
        for (int j = 0; j < 8; ++j) {
            float e = __expf(Sacc[j][r] - rmaxf[r]);
            Sacc[j][r] = e;
            s += e;
        }
        #pragma unroll
        for (int off = 1; off < 16; off <<= 1) s += __shfl_xor(s, off, 32);
        rsum[r] = s;
    }
    if (l16 == 0) {
        #pragma unroll
        for (int r = 0; r < 8; ++r) rsumW[wn][rowbase + r] = rsum[r];
    }
    __syncthreads();

    // ---- normalized P -> LDS (bf16, row-major) ----
    #pragma unroll
    for (int r = 0; r < 8; ++r) {
        float inv = 1.f / (rsumW[0][rowbase + r] + rsumW[1][rowbase + r]);
        #pragma unroll
        for (int j = 0; j < 8; ++j)
            sb.P[rowbase + r][n0 + j * 16 + l16] = (__bf16)(Sacc[j][r] * inv);
    }
    __syncthreads();

    // ---- GEMM2: U = P @ Q ----
    floatx8 Uacc[8];
    #pragma unroll
    for (int j = 0; j < 8; ++j)
        #pragma unroll
        for (int i = 0; i < 8; ++i) Uacc[j][i] = 0.f;

    const __bf16* prow = &sb.P[m0 + l16][0];
    for (int k0 = 0; k0 < TQ; k0 += 32) {
        const int kb = k0 + (hi ? 16 : 0);
        bf16x8 a0 = *(const bf16x8*)(prow + kb);
        bf16x8 a1 = *(const bf16x8*)(prow + kb + 8);
        bf16x16 a;
        #pragma unroll
        for (int i = 0; i < 8; ++i) { a[i] = a0[i]; a[i + 8] = a1[i]; }
        #pragma unroll
        for (int j = 0; j < 8; ++j) {
            const __bf16* qt = QT + ((size_t)b * DD + n0 + j * 16 + l16) * TQ
                                  + k0 + (hi ? 16 : 0);
            bf16x16 bf = *(const bf16x16*)qt;
            Uacc[j] = __builtin_amdgcn_wmma_f32_16x16x32_bf16(
                false, a, false, bf, (short)0, Uacc[j], false, false);
        }
    }

    // ---- epilogue: two 32-row halves, U via LDS, coalesced float4 stores ----
    for (int half = 0; half < 2; ++half) {
        __syncthreads();
        if ((wm >> 1) == half) {
            const int rb = rowbase - half * 32;
            #pragma unroll
            for (int j = 0; j < 8; ++j)
                #pragma unroll
                for (int r = 0; r < 8; ++r)
                    sb.U[rb + r][n0 + j * 16 + l16] = Uacc[j][r];
        }
        __syncthreads();
        #pragma unroll
        for (int it = 0; it < 8; ++it) {
            int idx  = it * 256 + tid;          // 0..2047 float4 slots
            int row  = idx >> 6;                // 0..31
            int col4 = (idx & 63) * 4;          // 0..252
            int t = t0 + half * 32 + row;
            float4 c4 = *(const float4*)(ctx + ((size_t)b * TC + t) * DD + col4);
            float4 u4 = *(const float4*)(&sb.U[row][col4]);
            float* gp = G + ((size_t)b * TC + t) * (4 * DD);
            *(float4*)(gp + col4)            = c4;
            *(float4*)(gp + DD + col4)       = u4;
            float4 cu;
            cu.x = c4.x * u4.x; cu.y = c4.y * u4.y;
            cu.z = c4.z * u4.z; cu.w = c4.w * u4.w;
            *(float4*)(gp + 2 * DD + col4)   = cu;
        }
    }
}

// ---------------------------------------------------------------------------
// Kernel 2: per batch: b_t = softmax_t(rowmax); h[d] = sum_t b_t * c[t,d]
// ---------------------------------------------------------------------------
__global__ __launch_bounds__(256)
void bt_h_kernel(const float* __restrict__ rowmax, const float* __restrict__ ctx,
                 float* __restrict__ h) {
    const int b = blockIdx.x;
    const int tid = threadIdx.x;
    __shared__ float p[TC];
    __shared__ float red[256];

    float m = -3.0e38f;
    for (int t = tid; t < TC; t += 256) {
        float v = rowmax[(size_t)b * TC + t];
        p[t] = v;
        m = fmaxf(m, v);
    }
    red[tid] = m;
    __syncthreads();
    for (int s = 128; s > 0; s >>= 1) {
        if (tid < s) red[tid] = fmaxf(red[tid], red[tid + s]);
        __syncthreads();
    }
    float mx = red[0];
    __syncthreads();

    float ssum = 0.f;
    for (int t = tid; t < TC; t += 256) {
        float e = __expf(p[t] - mx);
        p[t] = e;
        ssum += e;
    }
    red[tid] = ssum;
    __syncthreads();
    for (int s = 128; s > 0; s >>= 1) {
        if (tid < s) red[tid] += red[tid + s];
        __syncthreads();
    }
    float inv = 1.f / red[0];
    __syncthreads();

    float acc = 0.f;
    const float* cb = ctx + (size_t)b * TC * DD;
    for (int t = 0; t < TC; ++t)
        acc = fmaf(p[t], cb[(size_t)t * DD + tid], acc);
    h[b * DD + tid] = acc * inv;
}

// ---------------------------------------------------------------------------
// Kernel 3: G[:, :, 3D:4D] = C * h (broadcast over t), coalesced float4
// ---------------------------------------------------------------------------
__global__ __launch_bounds__(256)
void gmul_h_kernel(const float* __restrict__ ctx, const float* __restrict__ h,
                   float* __restrict__ G) {
    size_t idx = ((size_t)blockIdx.x * 256 + threadIdx.x) * 4;  // float index
    int d = (int)(idx & (DD - 1));
    size_t row = idx >> 8;          // b*TC + t
    int b = (int)(row >> 11);       // / TC
    float4 c4 = *(const float4*)(ctx + idx);
    float4 h4 = *(const float4*)(h + b * DD + d);
    float4 o;
    o.x = c4.x * h4.x; o.y = c4.y * h4.y;
    o.z = c4.z * h4.z; o.w = c4.w * h4.w;
    *(float4*)(G + row * (4 * DD) + 3 * DD + d) = o;
}

// ---------------------------------------------------------------------------
extern "C" void kernel_launch(void* const* d_in, const int* in_sizes, int n_in,
                              void* d_out, int out_size, void* d_ws, size_t ws_size,
                              hipStream_t stream) {
    const float* ctx   = (const float*)d_in[0];   // (B, TC, D)
    const float* query = (const float*)d_in[1];   // (B, TQ, D)
    const float* w     = (const float*)d_in[2];   // (3D,)
    float* G = (float*)d_out;                     // (B, TC, 4D)

    char* ws = (char*)d_ws;
    float* colconst = (float*)ws;                                 //  32 KB
    float* rowmax   = (float*)(ws + 32768);                       // 256 KB
    float* hbuf     = (float*)(ws + 32768 + 262144);              //  32 KB
    unsigned short* Qb = (unsigned short*)(ws + 327680);          //   4 MB
    unsigned short* QT = (unsigned short*)(ws + 327680 + 4194304);//   4 MB

    prep_q_kernel<<<BB * (TQ / 32), 256, 0, stream>>>(query, w, colconst, Qb, QT);

    dim3 g1(TC / 64, BB);
    attn_tile_kernel<<<g1, 256, 0, stream>>>(ctx, w, colconst, Qb, QT, rowmax, G);

    bt_h_kernel<<<BB, 256, 0, stream>>>(rowmax, ctx, hbuf);

    gmul_h_kernel<<<(BB * TC * DD / 4) / 256, 256, 0, stream>>>(ctx, hbuf, G);
}
